// TransformerTransducer_57166014709894
// MI455X (gfx1250) — matrix operand order repaired and checked by reading, not compiled, not run on hardware
//
#include <hip/hip_runtime.h>
#include <hip/hip_bf16.h>

// ---------------------------------------------------------------------------
// Problem constants (B,M,N,D,C) = (4,256,128,512,1024), JOINT = 2D = 1024
// ---------------------------------------------------------------------------
#define BB 4
#define MM 256
#define NN 128
#define DD 512
#define CC 1024

typedef __attribute__((ext_vector_type(16))) __bf16 v16bf;
typedef __attribute__((ext_vector_type(2)))  __bf16 v2bf;
typedef __attribute__((ext_vector_type(8)))  float  v8f;

union V16 {                 // one WMMA 16-bit operand (8 VGPRs / lane)
    v16bf          v;
    uint4          q[2];
    unsigned int   u[8];
    unsigned short h[16];
};
union F8 {                  // 8 consecutive f32
    float  f[8];
    float4 q[2];
};
union V2 {
    v2bf         v;
    unsigned int u;
};

__device__ __forceinline__ unsigned short f2bf(float f) {
    unsigned int u = __float_as_uint(f);
    unsigned int r = u + 0x7fffu + ((u >> 16) & 1u);   // round-to-nearest-even
    return (unsigned short)(r >> 16);
}

// hardware tanh (TRANS op v_tanh_f32 on gfx1250) with safe fallback
__device__ __forceinline__ float hw_tanh(float x) {
#if __has_builtin(__builtin_amdgcn_tanhf)
    return __builtin_amdgcn_tanhf(x);
#else
    return tanhf(x);
#endif
}

// packed f32x2 -> bf16x2 (v_cvt_pk_bf16_f32) with safe fallback
__device__ __forceinline__ unsigned int pack_bf16(float lo, float hi) {
#if __has_builtin(__builtin_amdgcn_cvt_pk_bf16_f32)
    V2 p;
    p.v = __builtin_amdgcn_cvt_pk_bf16_f32(lo, hi);
    return p.u;
#else
    return (unsigned int)f2bf(lo) | ((unsigned int)f2bf(hi) << 16);
#endif
}

// ---------------------------------------------------------------------------
// Stage 0: f32 -> bf16 conversion (2 elements / thread, packed store)
// ---------------------------------------------------------------------------
__global__ void cvt_bf16_kernel(const float* __restrict__ src,
                                unsigned int* __restrict__ dst, int n2) {
    int i = blockIdx.x * 256 + threadIdx.x;     // index over pairs
    if (i < n2) {
        const float2 f = ((const float2*)src)[i];
        dst[i] = pack_bf16(f.x, f.y);
    }
}

// ---------------------------------------------------------------------------
// Stage 1: fc1 via WMMA bf16.
//   he[bm, e] = sum_d enc[bm, d] * w1[e, d]       + b1[e]   (bm in [0,1024))
//   hd[bn, e] = sum_d dec[bn, d] * w1[e, 512 + d]           (bn in [0,512))
// One wave per (row-tile, e-tile). grid = (32 e-tiles, 96 row-tiles).
// ---------------------------------------------------------------------------
__global__ __launch_bounds__(32)
void fc1_kernel(const unsigned short* __restrict__ enc_bf,
                const unsigned short* __restrict__ dec_bf,
                const unsigned short* __restrict__ w1b,   // [512][1024] bf16
                const float* __restrict__ b1,
                float* __restrict__ he,                   // [1024][512]
                float* __restrict__ hd) {                 // [512][512]
    const int ct   = blockIdx.x;          // e tile 0..31
    const int rt   = blockIdx.y;          // row tile 0..95 (0..63 enc, 64..95 dec)
    const int lane = threadIdx.x & 31;
    const int li   = lane & 15;
    const int half = lane >> 4;

    const bool isEnc = (rt < 64);
    const int  r0    = isEnc ? rt * 16 : (rt - 64) * 16;
    const unsigned short* src  = isEnc ? enc_bf : dec_bf;
    const int  joff  = isEnc ? 0 : DD;
    float*     dst   = isEnc ? he : hd;

    const unsigned short* srow = src + (size_t)(r0 + li) * DD;

    // Documented bf16 A layout: lane<16 -> K = {base..base+7, base+16..base+23},
    // lane>=16 -> +8.  base = s*32 + half*8.
    V16 A[16];
#pragma unroll
    for (int s = 0; s < 16; ++s) {
        const int base = s * 32 + half * 8;
        A[s].q[0] = *(const uint4*)(srow + base);
        A[s].q[1] = *(const uint4*)(srow + base + 16);
    }

    // B tile: B[k, n] = w1[e = ct*16+n, joff + s*32 + k].
    const unsigned short* brow = w1b + (size_t)(ct * 16 + li) * (2 * DD) + joff + half * 8;

    v8f acc = {0.f, 0.f, 0.f, 0.f, 0.f, 0.f, 0.f, 0.f};
#pragma unroll
    for (int s = 0; s < 16; ++s) {
        V16 bv;
        bv.q[0] = *(const uint4*)(brow + s * 32);
        bv.q[1] = *(const uint4*)(brow + s * 32 + 16);
        acc = __builtin_amdgcn_wmma_f32_16x16x32_bf16(
            false, A[s].v, false, bv.v, (short)0, acc, false, false);
    }

    const float badd = isEnc ? b1[ct * 16 + li] : 0.0f;
#pragma unroll
    for (int r = 0; r < 8; ++r) {
        const int row = r0 + r + half * 8;
        dst[(size_t)row * DD + ct * 16 + li] = acc[r] + badd;
    }
}

// ---------------------------------------------------------------------------
// Stage 2: fused joint kernel, 64-row workgroups for 4x B-matrix L2 reuse.
// Workgroup = 8 waves (256 thr), owns 64 rows: fixed (b,m), n in [n0, n0+64).
//   - 8 waves cooperatively build the 64x512 bf16 tanh(he+hd) A tile in LDS
//     (each hid element computed exactly once), in WMMA A-operand layout.
//   - Wave w accumulates ALL 4 row-tiles x col-tiles [8w, 8w+8): each B tile
//     (w2 bf16, L2-resident) is fetched once and feeds 4 WMMAs.
//   - log-softmax: in-register partials + 16-lane shfl_xor + LDS cross-wave.
//   - output written once with non-temporal stores.
// L2 B traffic: 1 MB per 64 rows -> 2 GB total (vs 8 GB with 16-row tiles).
// ---------------------------------------------------------------------------
__global__ __launch_bounds__(256)
void joint_kernel(const float* __restrict__ he,            // [1024][512]
                  const float* __restrict__ hd,            // [512][512]
                  const unsigned short* __restrict__ w2b,  // [1024][512] bf16
                  const float* __restrict__ b2,            // [1024]
                  float* __restrict__ out) {               // [B,M,N,C]
    __shared__ unsigned short Ash[64 * 512];   // 64 KB: [(rt*16+s)][lane][16]
    __shared__ float redM[8][64];
    __shared__ float redS[8][64];

    const int t     = blockIdx.x;        // 0..2047
    const int nhalf = t & 1;             // N/64 = 2
    const int bm    = t >> 1;            // b*256 + m
    const int b     = bm >> 8;
    const int lane  = threadIdx.x & 31;
    const int wave  = threadIdx.x >> 5;  // 0..7
    const int li    = lane & 15;
    const int half  = lane >> 4;

    const float* heRow = he + (size_t)bm * DD;

    // ---- cooperative A build: wave w -> row-tile (w&3), K-slices 8*(w>>2).. ----
    {
        const int rt = wave & 3;
        const int sb = (wave >> 2) * 8;
        const float* hdRow =
            hd + (size_t)(b * NN + nhalf * 64 + rt * 16 + li) * DD;
#pragma unroll
        for (int q = 0; q < 8; ++q) {
            const int s    = sb + q;
            const int base = s * 32 + half * 8;
            F8 hc0, hc1, dc0, dc1;
            hc0.q[0] = *(const float4*)(heRow + base);
            hc0.q[1] = *(const float4*)(heRow + base + 4);
            hc1.q[0] = *(const float4*)(heRow + base + 16);
            hc1.q[1] = *(const float4*)(heRow + base + 20);
            dc0.q[0] = *(const float4*)(hdRow + base);
            dc0.q[1] = *(const float4*)(hdRow + base + 4);
            dc1.q[0] = *(const float4*)(hdRow + base + 16);
            dc1.q[1] = *(const float4*)(hdRow + base + 20);
            V16 av;
#pragma unroll
            for (int j = 0; j < 4; ++j) {
                av.u[j]     = pack_bf16(hw_tanh(hc0.f[2*j]   + dc0.f[2*j]),
                                        hw_tanh(hc0.f[2*j+1] + dc0.f[2*j+1]));
                av.u[j + 4] = pack_bf16(hw_tanh(hc1.f[2*j]   + dc1.f[2*j]),
                                        hw_tanh(hc1.f[2*j+1] + dc1.f[2*j+1]));
            }
            *(uint4*)&Ash[(rt * 16 + s) * 512 + lane * 16]     = av.q[0];
            *(uint4*)&Ash[(rt * 16 + s) * 512 + lane * 16 + 8] = av.q[1];
        }
    }
    __syncthreads();

    // ---- GEMM: acc[rt][ci], wave handles col tiles [wave*8, wave*8+8) ----
    v8f acc[4][8];
#pragma unroll
    for (int rt = 0; rt < 4; ++rt)
#pragma unroll
        for (int ci = 0; ci < 8; ++ci)
            acc[rt][ci] = (v8f){0.f, 0.f, 0.f, 0.f, 0.f, 0.f, 0.f, 0.f};

#pragma unroll
    for (int s = 0; s < 16; ++s) {
        V16 As[4];
#pragma unroll
        for (int rt = 0; rt < 4; ++rt) {
            As[rt].q[0] = *(const uint4*)&Ash[(rt * 16 + s) * 512 + lane * 16];
            As[rt].q[1] = *(const uint4*)&Ash[(rt * 16 + s) * 512 + lane * 16 + 8];
        }
#pragma unroll
        for (int ci = 0; ci < 8; ++ci) {
            const int c = wave * 8 + ci;
            const unsigned short* brow =
                w2b + (size_t)(c * 16 + li) * DD + half * 8 + s * 32;
            V16 bv;
            bv.q[0] = *(const uint4*)(brow);
            bv.q[1] = *(const uint4*)(brow + 16);
#pragma unroll
            for (int rt = 0; rt < 4; ++rt)
                acc[rt][ci] = __builtin_amdgcn_wmma_f32_16x16x32_bf16(
                    false, As[rt].v, false, bv.v, (short)0, acc[rt][ci],
                    false, false);
        }
    }

    // ---- + b2 ----
#pragma unroll
    for (int ci = 0; ci < 8; ++ci) {
        const float b2v = b2[(wave * 8 + ci) * 16 + li];
#pragma unroll
        for (int rt = 0; rt < 4; ++rt)
#pragma unroll
            for (int r = 0; r < 8; ++r) acc[rt][ci][r] += b2v;
    }

    // ---- per-(row-tile) partial max / sum-exp over this wave's 128 cols ----
#pragma unroll
    for (int rt = 0; rt < 4; ++rt) {
        float pmax[8], psum[8];
#pragma unroll
        for (int r = 0; r < 8; ++r) {
            float mx = acc[rt][0][r];
#pragma unroll
            for (int ci = 1; ci < 8; ++ci) mx = fmaxf(mx, acc[rt][ci][r]);
            pmax[r] = mx;
        }
#pragma unroll
        for (int off = 1; off < 16; off <<= 1)
#pragma unroll
            for (int r = 0; r < 8; ++r)
                pmax[r] = fmaxf(pmax[r], __shfl_xor(pmax[r], off, 32));
#pragma unroll
        for (int r = 0; r < 8; ++r) {
            float s = 0.f;
#pragma unroll
            for (int ci = 0; ci < 8; ++ci)
                s += __expf(acc[rt][ci][r] - pmax[r]);
            psum[r] = s;
        }
#pragma unroll
        for (int off = 1; off < 16; off <<= 1)
#pragma unroll
            for (int r = 0; r < 8; ++r)
                psum[r] += __shfl_xor(psum[r], off, 32);
        if (li == 0) {
#pragma unroll
            for (int r = 0; r < 8; ++r) {
                const int row = rt * 16 + half * 8 + r;
                redM[wave][row] = pmax[r];
                redS[wave][row] = psum[r];
            }
        }
    }
    __syncthreads();

    // ---- cross-wave combine -> lse per row ----
    float lse[4][8];
#pragma unroll
    for (int rt = 0; rt < 4; ++rt)
#pragma unroll
        for (int r = 0; r < 8; ++r) {
            const int row = rt * 16 + half * 8 + r;
            float gmax = redM[0][row];
#pragma unroll
            for (int w = 1; w < 8; ++w) gmax = fmaxf(gmax, redM[w][row]);
            float gsum = 0.f;
#pragma unroll
            for (int w = 0; w < 8; ++w)
                gsum += redS[w][row] * __expf(redM[w][row] - gmax);
            lse[rt][r] = gmax + __logf(gsum);
        }

    // ---- streamed output: log_softmax = logits - lse ----
    const size_t rowBase = ((size_t)bm * NN + nhalf * 64) * CC;
#pragma unroll
    for (int rt = 0; rt < 4; ++rt)
#pragma unroll
        for (int ci = 0; ci < 8; ++ci) {
            const int c = wave * 8 + ci;
#pragma unroll
            for (int r = 0; r < 8; ++r) {
                const int    row = rt * 16 + half * 8 + r;
                const size_t idx = rowBase + (size_t)row * CC + c * 16 + li;
                __builtin_nontemporal_store(acc[rt][ci][r] - lse[rt][r],
                                            &out[idx]);
            }
        }
}

// ---------------------------------------------------------------------------
// Host launcher
// ---------------------------------------------------------------------------
extern "C" void kernel_launch(void* const* d_in, const int* in_sizes, int n_in,
                              void* d_out, int out_size, void* d_ws, size_t ws_size,
                              hipStream_t stream) {
    const float* enc = (const float*)d_in[0];   // [4,256,512]  = 524288
    const float* dec = (const float*)d_in[1];   // [4,128,512]  = 262144
    const float* w1  = (const float*)d_in[2];   // [512,1024]   = 524288
    const float* b1  = (const float*)d_in[3];   // [512]
    const float* w2  = (const float*)d_in[4];   // [1024,512]   = 524288
    const float* b2  = (const float*)d_in[5];   // [1024]
    float* out = (float*)d_out;

    const int nEnc = BB * MM * DD;   // 524288
    const int nDec = BB * NN * DD;   // 262144
    const int nW1  = DD * 2 * DD;    // 524288
    const int nW2  = CC * DD;        // 524288

    unsigned short* enc_bf = (unsigned short*)d_ws;
    unsigned short* dec_bf = enc_bf + nEnc;
    unsigned short* w1b    = dec_bf + nDec;
    unsigned short* w2b    = w1b + nW1;
    float* he = (float*)(w2b + nW2);          // [1024][512] f32
    float* hd = he + (size_t)BB * MM * DD;    // [512][512]  f32
    // total workspace: 3.5 MB bf16 + 3 MB f32 = 6.5 MB

    cvt_bf16_kernel<<<(nEnc / 2 + 255) / 256, 256, 0, stream>>>(
        enc, (unsigned int*)enc_bf, nEnc / 2);
    cvt_bf16_kernel<<<(nDec / 2 + 255) / 256, 256, 0, stream>>>(
        dec, (unsigned int*)dec_bf, nDec / 2);
    cvt_bf16_kernel<<<(nW1 / 2 + 255) / 256, 256, 0, stream>>>(
        w1, (unsigned int*)w1b, nW1 / 2);
    cvt_bf16_kernel<<<(nW2 / 2 + 255) / 256, 256, 0, stream>>>(
        w2, (unsigned int*)w2b, nW2 / 2);

    dim3 g1(DD / 16, (BB * MM + BB * NN) / 16);   // (32, 96)
    fc1_kernel<<<g1, 32, 0, stream>>>(enc_bf, dec_bf, w1b, b1, he, hd);

    const int nTiles = BB * MM * (NN / 64);       // 2048 workgroups x 256 thr
    joint_kernel<<<nTiles, 256, 0, stream>>>(he, hd, w2b, b2, out);
}
